// State_encoder_26714696581521
// MI455X (gfx1250) — matrix operand order repaired and checked by reading, not compiled, not run on hardware
//
#include <hip/hip_runtime.h>
#include <hip/hip_bf16.h>

// ---------------- problem constants ----------------
#define BB     1024
#define NN     62
#define FI     5
#define NF     310          // N*F_IN
#define NN2    3844         // N*N
#define KP     3872         // N*N padded to multiple of 32 (121*32)
#define ED     1024         // EMB_DIM
#define NE     64           // NUM_EMB
#define ROWS   5120         // B*F_IN
#define NUP    3904         // NN2 padded to multiple of 64 (61*64)
#define X2SZ   (BB*NN*50)   // 3,174,400

typedef __attribute__((ext_vector_type(16))) __bf16 v16bf;
typedef __attribute__((ext_vector_type(8)))  float  v8f;

union FragBF { v16bf v; unsigned int u[8]; };
union FragF  { v8f v; float f[8]; };

__device__ __forceinline__ float dev_elu(float v)  { return v > 0.f ? v : __expf(v) - 1.f; }
__device__ __forceinline__ float dev_selu(float v) {
    const float l = 1.0507009873554805f, a = 1.6732632423543772f;
    return v > 0.f ? l * v : l * a * (__expf(v) - 1.f);
}
__device__ __forceinline__ float dev_sig(float v)  { return 1.f / (1.f + __expf(-v)); }

// ============================================================
// WMMA bf16 GEMM:  C(MxN) = A(MxK) * B^T-stored(NxK), f32 accum
// block = 128 threads (4 waves); each wave computes a 32x64 tile
// (2 M-frags x 4 N-frags = 8 wmma per k-step) -> block tile 128x64
// ============================================================
__global__ __launch_bounds__(128)
void k_gemm_bf16(const __bf16* __restrict__ A, int lda,
                 const __bf16* __restrict__ B, int ldb,
                 float* __restrict__ C, int ldc,
                 __bf16* __restrict__ Cbf,
                 int ksteps, int nout, int mmax)
{
    const int lane = threadIdx.x & 31;
    const int wave = threadIdx.x >> 5;
    const int h    = lane >> 4;       // K-half selector
    const int l16  = lane & 15;
    const int m0   = blockIdx.x * 128 + wave * 32;
    const int n0   = blockIdx.y * 64;

    // clamp A row indices so partial-M launches never read far out of range
    const int r0 = min(m0 + l16,      mmax - 1);
    const int r1 = min(m0 + 16 + l16, mmax - 1);
    const __bf16* __restrict__ arow0 = A + (size_t)r0 * lda;
    const __bf16* __restrict__ arow1 = A + (size_t)r1 * lda;

    FragF acc[2][4];
#pragma unroll
    for (int g = 0; g < 2; ++g)
#pragma unroll
        for (int s = 0; s < 4; ++s)
            acc[g][s].v = v8f{0.f,0.f,0.f,0.f,0.f,0.f,0.f,0.f};

    for (int ks = 0; ks < ksteps; ++ks) {
        const int k0 = ks * 32;
        FragBF a0, a1;
#pragma unroll
        for (int v = 0; v < 8; ++v) {
            const int kp = (v < 4) ? (8 * h + 2 * v) : (16 + 8 * h + 2 * (v - 4));
            a0.u[v] = *(const unsigned int*)(arow0 + k0 + kp);
            a1.u[v] = *(const unsigned int*)(arow1 + k0 + kp);
        }
        __builtin_prefetch(arow0 + k0 + 32, 0, 1);
        __builtin_prefetch(arow1 + k0 + 32, 0, 1);
#pragma unroll
        for (int s = 0; s < 4; ++s) {
            const __bf16* __restrict__ brow =
                B + (size_t)(n0 + s * 16 + l16) * ldb + k0;
            FragBF bfr;
#pragma unroll
            for (int v = 0; v < 8; ++v)
                bfr.u[v] = *(const unsigned int*)(brow + ((v < 4) ? (8 * h + 2 * v) : (16 + 8 * h + 2 * (v - 4))));
            acc[0][s].v = __builtin_amdgcn_wmma_f32_16x16x32_bf16(
                false, a0.v, false, bfr.v, (short)0, acc[0][s].v, false, false);
            acc[1][s].v = __builtin_amdgcn_wmma_f32_16x16x32_bf16(
                false, a1.v, false, bfr.v, (short)0, acc[1][s].v, false, false);
        }
    }
#pragma unroll
    for (int g = 0; g < 2; ++g) {
#pragma unroll
        for (int s = 0; s < 4; ++s) {
            const int n = n0 + s * 16 + l16;
            if (n < nout) {
#pragma unroll
                for (int r = 0; r < 8; ++r) {
                    const int m = m0 + g * 16 + r + 8 * h;
                    if (m < mmax) {
                        const float val = acc[g][s].f[r];
                        C[(size_t)m * ldc + n] = val;
                        if (Cbf) Cbf[(size_t)m * ldc + n] = (__bf16)val;
                    }
                }
            }
        }
    }
}

// ============================================================
// get_adj: o = p@x + bias ; g = elu((o@q)@theta) ; normalize;
// emit bf16 rows of z into zA (K padded to 3872, zero tail)
// ============================================================
__global__ __launch_bounds__(256)
void k_get_adj(const float* __restrict__ x, const float* __restrict__ p,
               const float* __restrict__ bias, const float* __restrict__ q,
               const float* __restrict__ theta, __bf16* __restrict__ zA)
{
    __shared__ float  xs[NF], os[NF], ts[NF], dsum[NF];
    __shared__ __bf16 gs[NN * NF];            // 62*310 bf16 = 38.4 KB
    const int b = blockIdx.x, t = threadIdx.x;

    if (t < NF) xs[t] = x[(size_t)b * NF + t];
    __syncthreads();
    if (t < NF) {
        const int i = t / FI, d = t % FI;
        float s = 0.f;
        for (int j = 0; j < NN; ++j) s += p[i * NN + j] * xs[j * FI + d];
        os[t] = s + bias[t];
    }
    __syncthreads();
    if (t < NF) {
        const int i = t / FI, d = t % FI;
        float s = 0.f;
        for (int k = 0; k < FI; ++k) s += os[i * FI + k] * q[k * FI + d];
        ts[t] = s;
    }
    __syncthreads();
    for (int c = t; c < NN * NF; c += 256) {
        const int i = c / NF, col = c % NF;
        float s = 0.f;
        for (int k = 0; k < FI; ++k) s += ts[i * FI + k] * theta[k * NF + col];
        gs[c] = (__bf16)dev_elu(s);
    }
    __syncthreads();
    if (t < NF) {                              // t -> (f, i)
        const int f = t / NN, i = t % NN;
        float s = 0.f;
        for (int j = 0; j < NN; ++j) s += (float)gs[i * NF + j * FI + f];
        dsum[t] = rsqrtf(fabsf(s) + 1e-6f);
    }
    __syncthreads();
    for (int c = t; c < FI * NN2; c += 256) {
        const int f = c / NN2, rem = c % NN2, i = rem / NN, j = rem % NN;
        const float v =
            (float)gs[i * NF + j * FI + f] * dsum[f * NN + i] * dsum[f * NN + j];
        zA[(size_t)(b * FI + f) * KP + rem] = (__bf16)v;
    }
    if (t < FI * (KP - NN2)) {
        const int f = t / (KP - NN2), k = NN2 + t % (KP - NN2);
        zA[(size_t)(b * FI + f) * KP + k] = (__bf16)0.f;
    }
}

// ---------------- weight conversions ----------------
__global__ void k_conv_wd(const float* __restrict__ wd, __bf16* __restrict__ wdT) {
    const int i = blockIdx.x * blockDim.x + threadIdx.x;     // over 1024*3872
    if (i >= ED * KP) return;
    const int n = i / KP, k = i % KP;
    wdT[i] = (k < NN2) ? (__bf16)wd[(size_t)k * ED + n] : (__bf16)0.f;
}
__global__ void k_conv_wu(const float* __restrict__ wu, __bf16* __restrict__ wuT) {
    const int i = blockIdx.x * blockDim.x + threadIdx.x;     // over 3904*1024
    if (i >= NUP * ED) return;
    const int n = i / ED, k = i % ED;
    wuT[i] = (n < NN2) ? (__bf16)wu[(size_t)k * NN2 + n] : (__bf16)0.f;
}
__global__ void k_conv_embed(const float* __restrict__ e, __bf16* __restrict__ ebf) {
    const int i = blockIdx.x * blockDim.x + threadIdx.x;     // over 65536
    if (i < NE * ED) ebf[i] = (__bf16)e[i];
}
__global__ void k_enorm(const float* __restrict__ embed, float* __restrict__ enorm) {
    __shared__ float red[256];
    const int j = blockIdx.x;
    float s = 0.f;
    for (int k = threadIdx.x; k < ED; k += 256) {
        const float v = embed[(size_t)j * ED + k];
        s += v * v;
    }
    red[threadIdx.x] = s;
    __syncthreads();
    for (int o = 128; o > 0; o >>= 1) {
        if (threadIdx.x < o) red[threadIdx.x] += red[threadIdx.x + o];
        __syncthreads();
    }
    if (threadIdx.x == 0) enorm[j] = red[0];
}

__global__ void k_zero(float* __restrict__ vq, float* __restrict__ usage) {
    if (threadIdx.x == 0) vq[0] = 0.f;
    if (threadIdx.x < NE) usage[threadIdx.x] = 0.f;
}

// ---------------- argmin / VQ loss / usage ----------------
__global__ __launch_bounds__(256)
void k_argmin(const float* __restrict__ zf, const float* __restrict__ dots,
              const float* __restrict__ enorm, int* __restrict__ idx,
              float* __restrict__ vq_accum, float* __restrict__ usage)
{
    __shared__ float red[256];
    __shared__ float d2s[NE];
    const int r = blockIdx.x, t = threadIdx.x;
    float s = 0.f;
    for (int k = t; k < ED; k += 256) {
        const float v = zf[(size_t)r * ED + k];
        s += v * v;
    }
    red[t] = s;
    __syncthreads();
    for (int o = 128; o > 0; o >>= 1) {
        if (t < o) red[t] += red[t + o];
        __syncthreads();
    }
    if (t < NE) d2s[t] = red[0] - 2.f * dots[(size_t)r * NE + t] + enorm[t];
    __syncthreads();
    if (t == 0) {
        int best = 0;
        float bv = d2s[0];
        for (int j = 1; j < NE; ++j)
            if (d2s[j] < bv) { bv = d2s[j]; best = j; }
        idx[r] = best;
        atomicAdd(vq_accum, bv);
        atomicAdd(&usage[best], 1.f);
    }
}

// ---------------- recon gather + SE + renormalize ----------------
__global__ __launch_bounds__(256)
void k_recon_se(const float* __restrict__ eu, const int* __restrict__ idx,
                const float* __restrict__ sw1, const float* __restrict__ sb1,
                const float* __restrict__ sw2, const float* __restrict__ sb2,
                float* __restrict__ adjr)
{
    __shared__ float T[NN2];
    __shared__ float red[FI * 256];
    __shared__ float ga[FI];
    __shared__ int   id5[FI];
    __shared__ float dinv[NN];
    const int b = blockIdx.x, t = threadIdx.x;

    if (t < FI) id5[t] = idx[b * FI + t];
    __syncthreads();
    float acc[FI] = {0.f, 0.f, 0.f, 0.f, 0.f};
    for (int c = t; c < NN2; c += 256)
        for (int f = 0; f < FI; ++f) acc[f] += eu[(size_t)id5[f] * NN2 + c];
    for (int f = 0; f < FI; ++f) red[f * 256 + t] = acc[f];
    __syncthreads();
    for (int o = 128; o > 0; o >>= 1) {
        if (t < o)
            for (int f = 0; f < FI; ++f) red[f * 256 + t] += red[f * 256 + t + o];
        __syncthreads();
    }
    if (t == 0) {
        float h = sb1[0];
        for (int f = 0; f < FI; ++f) h += (red[f * 256] / (float)NN2) * sw1[f];
        h = fmaxf(h, 0.f);
        for (int f = 0; f < FI; ++f) ga[f] = 1.f + dev_sig(h * sw2[f] + sb2[f]);
    }
    __syncthreads();
    for (int c = t; c < NN2; c += 256) {
        float s = 0.f;
        for (int f = 0; f < FI; ++f)
            s += dev_elu(eu[(size_t)id5[f] * NN2 + c] * ga[f]);
        T[c] = dev_elu(s);
    }
    __syncthreads();
    if (t < NN) {
        float s = 0.f;
        for (int j = 0; j < NN; ++j) s += T[t * NN + j];
        dinv[t] = rsqrtf(fabsf(s) + 1e-6f);
    }
    __syncthreads();
    for (int c = t; c < NN2; c += 256) {
        const int i = c / NN, j = c % NN;
        adjr[(size_t)b * NN2 + c] = T[c] * dinv[i] * dinv[j];
    }
}

// ---------------- CBAM on an LDS buffer (62 x D), in place ----------------
__device__ void cbam_inplace(float* xb, int D,
                             const float* __restrict__ w1, const float* __restrict__ b1,
                             const float* __restrict__ w2, const float* __restrict__ b2,
                             const float* __restrict__ conv,
                             float* chm, float* chx, float* hm, float* hx,
                             float* ca, float* spm, float* spx, float* sa)
{
    const int t = threadIdx.x;
    __syncthreads();
    if (t < NN) {
        float m = 0.f, mx = -1e30f;
        for (int d = 0; d < D; ++d) {
            const float v = xb[t * D + d];
            m += v; mx = fmaxf(mx, v);
        }
        chm[t] = m / (float)D; chx[t] = mx;
    }
    __syncthreads();
    if (t < 3) {
        float s1 = b1[t], s2 = b1[t];
        for (int i = 0; i < NN; ++i) {
            s1 += chm[i] * w1[i * 3 + t];
            s2 += chx[i] * w1[i * 3 + t];
        }
        hm[t] = fmaxf(s1, 0.f); hx[t] = fmaxf(s2, 0.f);
    }
    __syncthreads();
    if (t < NN) {
        float s1 = b2[t], s2 = b2[t];
        for (int k = 0; k < 3; ++k) {
            s1 += hm[k] * w2[k * NN + t];
            s2 += hx[k] * w2[k * NN + t];
        }
        ca[t] = dev_sig(s1 + s2);
    }
    __syncthreads();
    for (int c = t; c < NN * D; c += 256) xb[c] *= ca[c / D];
    __syncthreads();
    if (t < D) {
        float m = 0.f, mx = -1e30f;
        for (int i = 0; i < NN; ++i) {
            const float v = xb[i * D + t];
            m += v; mx = fmaxf(mx, v);
        }
        spm[t] = m / (float)NN; spx[t] = mx;
    }
    __syncthreads();
    if (t < D) {
        float s = 0.f;
        for (int k = 0; k < 7; ++k) {
            const int dd = t - 3 + k;
            if (dd >= 0 && dd < D) s += conv[k] * spm[dd] + conv[7 + k] * spx[dd];
        }
        sa[t] = dev_sig(s);
    }
    __syncthreads();
    for (int c = t; c < NN * D; c += 256) xb[c] *= sa[c % D];
    __syncthreads();
}

// ---------------- res_gcn: two GCN layers + two CBAMs ----------------
__global__ __launch_bounds__(256)
void k_gcn(const float* __restrict__ adjr, const float* __restrict__ x,
           const float* __restrict__ w1, const float* __restrict__ w2,
           const float* __restrict__ c1w1, const float* __restrict__ c1b1,
           const float* __restrict__ c1w2, const float* __restrict__ c1b2,
           const float* __restrict__ c1conv,
           const float* __restrict__ c2w1, const float* __restrict__ c2b1,
           const float* __restrict__ c2w2, const float* __restrict__ c2b2,
           const float* __restrict__ c2conv,
           float* __restrict__ out)
{
    __shared__ float As[NN2], xs[NF], xw[NN * 30], bc1[NN * 20], bc2[NN * 50];
    __shared__ float chm[NN], chx[NN], hm[3], hx[3], ca[NN], spm[50], spx[50], sa[50];
    const int b = blockIdx.x, t = threadIdx.x;

    for (int c = t; c < NN2; c += 256) As[c] = adjr[(size_t)b * NN2 + c];
    if (t < NF) xs[t] = x[(size_t)b * NF + t];
    __syncthreads();
    // xw1 = x @ w1   (62 x 15)
    for (int c = t; c < NN * 15; c += 256) {
        const int i = c / 15, d = c % 15;
        float s = 0.f;
        for (int k = 0; k < FI; ++k) s += xs[i * FI + k] * w1[k * 15 + d];
        xw[c] = s;
    }
    __syncthreads();
    // bc1 = concat(x, selu(A @ xw1))   (62 x 20)
    for (int c = t; c < NN * 15; c += 256) {
        const int i = c / 15, d = c % 15;
        float s = 0.f;
        for (int j = 0; j < NN; ++j) s += As[i * NN + j] * xw[j * 15 + d];
        bc1[i * 20 + 5 + d] = dev_selu(s);
    }
    for (int c = t; c < NF; c += 256) bc1[(c / FI) * 20 + (c % FI)] = xs[c];
    __syncthreads();
    cbam_inplace(bc1, 20, c1w1, c1b1, c1w2, c1b2, c1conv,
                 chm, chx, hm, hx, ca, spm, spx, sa);
    // xw2 = bc1 @ w2   (62 x 30)
    for (int c = t; c < NN * 30; c += 256) {
        const int i = c / 30, d = c % 30;
        float s = 0.f;
        for (int k = 0; k < 20; ++k) s += bc1[i * 20 + k] * w2[k * 30 + d];
        xw[c] = s;
    }
    __syncthreads();
    // bc2 = concat(bc1, selu(A @ xw2))   (62 x 50)
    for (int c = t; c < NN * 30; c += 256) {
        const int i = c / 30, d = c % 30;
        float s = 0.f;
        for (int j = 0; j < NN; ++j) s += As[i * NN + j] * xw[j * 30 + d];
        bc2[i * 50 + 20 + d] = dev_selu(s);
    }
    for (int c = t; c < NN * 20; c += 256) bc2[(c / 20) * 50 + (c % 20)] = bc1[c];
    __syncthreads();
    cbam_inplace(bc2, 50, c2w1, c2b1, c2w2, c2b2, c2conv,
                 chm, chx, hm, hx, ca, spm, spx, sa);
    for (int c = t; c < NN * 50; c += 256)
        out[1 + (size_t)b * (NN * 50) + c] = bc2[c];
}

// ---------------- finalize: vq_loss, usage, embed passthrough ----------------
__global__ void k_finalize(const float* __restrict__ vq_accum,
                           const float* __restrict__ usage_cnt,
                           const float* __restrict__ embed,
                           float* __restrict__ out)
{
    const int t = blockIdx.x * blockDim.x + threadIdx.x;
    if (t == 0) out[0] = 1.25f * vq_accum[0] / ((float)ROWS * (float)ED);
    if (t < NE) out[1 + X2SZ + t] = usage_cnt[t] / (float)ROWS;
    if (t < NE * ED) out[1 + X2SZ + NE + t] = embed[t];
}

// ============================================================
extern "C" void kernel_launch(void* const* d_in, const int* in_sizes, int n_in,
                              void* d_out, int out_size, void* d_ws, size_t ws_size,
                              hipStream_t stream)
{
    const float* x      = (const float*)d_in[0];
    const float* p      = (const float*)d_in[1];
    const float* bias   = (const float*)d_in[2];
    const float* q      = (const float*)d_in[3];
    const float* theta  = (const float*)d_in[4];
    const float* se_w1  = (const float*)d_in[5];
    const float* se_b1  = (const float*)d_in[6];
    const float* se_w2  = (const float*)d_in[7];
    const float* se_b2  = (const float*)d_in[8];
    const float* cb_wd  = (const float*)d_in[9];
    const float* cb_wu  = (const float*)d_in[10];
    const float* embed  = (const float*)d_in[11];
    const float* w1     = (const float*)d_in[12];
    const float* w2     = (const float*)d_in[13];
    const float* c1w1   = (const float*)d_in[14];
    const float* c1b1   = (const float*)d_in[15];
    const float* c1w2   = (const float*)d_in[16];
    const float* c1b2   = (const float*)d_in[17];
    const float* c1conv = (const float*)d_in[18];
    const float* c2w1   = (const float*)d_in[19];
    const float* c2b1   = (const float*)d_in[20];
    const float* c2w2   = (const float*)d_in[21];
    const float* c2b2   = (const float*)d_in[22];
    const float* c2conv = (const float*)d_in[23];
    float* out = (float*)d_out;

    // ------- workspace carve-out -------
    char* ws = (char*)d_ws;
    size_t off = 0;
    auto carve = [&](size_t bytes) -> char* {
        char* r = ws + off;
        off = (off + bytes + 255) & ~(size_t)255;
        return r;
    };
    __bf16* zA    = (__bf16*)carve((size_t)ROWS * KP * 2);     // 39.6 MB
    __bf16* wdT   = (__bf16*)carve((size_t)ED * KP * 2);       //  7.9 MB
    __bf16* wuT   = (__bf16*)carve((size_t)NUP * ED * 2);      //  8.0 MB
    __bf16* ebf   = (__bf16*)carve((size_t)NE * ED * 2);       //  0.1 MB
    float*  zf    = (float*) carve((size_t)ROWS * ED * 4);     // 21.0 MB
    __bf16* zfb   = (__bf16*)carve((size_t)ROWS * ED * 2);     // 10.5 MB
    float*  dots  = (float*) carve((size_t)ROWS * NE * 4);     //  1.3 MB
    float*  enorm = (float*) carve((size_t)NE * 4);
    float*  eu    = (float*) carve((size_t)NE * NN2 * 4);      //  1.0 MB
    float*  adjr  = (float*) carve((size_t)BB * NN2 * 4);      // 15.7 MB
    int*    idx   = (int*)   carve((size_t)ROWS * 4);
    float*  vqacc = (float*) carve(4);
    float*  usage = (float*) carve(NE * 4);
    (void)ws_size; (void)in_sizes; (void)n_in; (void)out_size;

    // 1) adjacency generation -> bf16 z rows
    k_get_adj<<<BB, 256, 0, stream>>>(x, p, bias, q, theta, zA);
    // 2) weight / codebook conversions
    k_conv_wd<<<(ED * KP + 255) / 256, 256, 0, stream>>>(cb_wd, wdT);
    k_conv_wu<<<(NUP * ED + 255) / 256, 256, 0, stream>>>(cb_wu, wuT);
    k_conv_embed<<<(NE * ED + 255) / 256, 256, 0, stream>>>(embed, ebf);
    k_enorm<<<NE, 256, 0, stream>>>(embed, enorm);
    // 3) zf = z @ cb_wd        (5120 x 1024, K = 3872)   [WMMA]
    k_gemm_bf16<<<dim3(ROWS / 128, ED / 64), 128, 0, stream>>>(
        zA, KP, wdT, KP, zf, ED, zfb, KP / 32, ED, ROWS);
    // 4) dots = zf @ embed^T   (5120 x 64, K = 1024)     [WMMA]
    k_gemm_bf16<<<dim3(ROWS / 128, 1), 128, 0, stream>>>(
        zfb, ED, ebf, ED, dots, NE, nullptr, ED / 32, NE, ROWS);
    // 5) embed_up = embed @ cb_wu  (64 x 3844, K = 1024) [WMMA]
    k_gemm_bf16<<<dim3(1, NUP / 64), 128, 0, stream>>>(
        ebf, ED, wuT, ED, eu, NN2, nullptr, ED / 32, NN2, NE);
    // 6) VQ argmin + loss + usage
    k_zero<<<1, 64, 0, stream>>>(vqacc, usage);
    k_argmin<<<ROWS, 256, 0, stream>>>(zf, dots, enorm, idx, vqacc, usage);
    // 7) recon gather + SE + renormalize
    k_recon_se<<<BB, 256, 0, stream>>>(eu, idx, se_w1, se_b1, se_w2, se_b2, adjr);
    // 8) res_gcn + CBAMs -> x2
    k_gcn<<<BB, 256, 0, stream>>>(adjr, x, w1, w2,
                                  c1w1, c1b1, c1w2, c1b2, c1conv,
                                  c2w1, c2b1, c2w2, c2b2, c2conv, out);
    // 9) scalars + passthrough outputs
    k_finalize<<<(NE * ED + 255) / 256, 256, 0, stream>>>(vqacc, usage, embed, out);
}
